// QwenAttention_heavy_hitter_62165356642726
// MI455X (gfx1250) — matrix operand order repaired and checked by reading, not compiled, not run on hardware
//
#include <hip/hip_runtime.h>
#include <math.h>

// ---------------------------------------------------------------- constants
#define Hn    32
#define KVH   8
#define Dd    128
#define HID   4096      // Hn * Dd
#define Ss    2048
#define KVLD  1024      // KVH * Dd
#define HEAVY 204       // int(0.1 * 2048)
#define RECENT 204

#define SC_STRIDE 2052  // 2048 + 4 pad (floats)

// ---------------------------------------------------------------- types
typedef __attribute__((ext_vector_type(16))) __bf16          v16bf;
typedef __attribute__((ext_vector_type(8)))  float           v8f;
typedef __attribute__((ext_vector_type(8)))  unsigned short  u16x8;
typedef __attribute__((ext_vector_type(4)))  float           f32x4;

union FragAB {
    u16x8          h[2];
    unsigned short s[16];
    v16bf          v;
};
static_assert(sizeof(FragAB) == 32, "frag size");

__device__ __forceinline__ unsigned short f2bf(float f) {
    union { float f; unsigned int u; } c; c.f = f;
    unsigned int u = c.u;
    unsigned int r = (u + 0x7fffu + ((u >> 16) & 1u)) >> 16;   // RNE
    return (unsigned short)r;
}

__device__ __forceinline__ v8f vzero8() {
    v8f z = {0.f,0.f,0.f,0.f,0.f,0.f,0.f,0.f};
    return z;
}

__device__ __forceinline__ v8f wmma_bf16(const FragAB& a, const FragAB& b, v8f c) {
    return __builtin_amdgcn_wmma_f32_16x16x32_bf16(
        /*neg_a=*/false, a.v, /*neg_b=*/false, b.v,
        /*c_mod=*/(short)0, c, /*reuse_a=*/false, /*reuse_b=*/false);
}

// ---- CDNA5 async global->LDS copy (ASYNCcnt path, CDNA5 ISA)
// 16 bytes per lane: global data at vaddr -> LDS at vdst.  Low 32 bits of a generic LDS
// pointer are the LDS byte offset.
__device__ __forceinline__ void async_copy_b128(const void* g, void* lds_ptr) {
    unsigned           ldsOff = (unsigned)(size_t)lds_ptr;
    unsigned long long ga     = (unsigned long long)(size_t)g;
    asm volatile("global_load_async_to_lds_b128 %0, %1, off"
                 :: "v"(ldsOff), "v"(ga) : "memory");
}
__device__ __forceinline__ void wait_async_zero() {
    asm volatile("s_wait_asynccnt 0" ::: "memory");
}

// ================================================================ GEMM
// C[M,N] = scale * (A[M,K] @ W[K,N] + bias)
// A: fp32 or bf16(ushort) row-major.  W: fp32 row-major (converted to bf16
// while staging).  Output: bf16(ushort) or fp32, row-major (ld = N).
// bf16 A tiles are staged with GLOBAL_LOAD_ASYNC_TO_LDS_B128.
template<bool A_BF16, bool OUT_BF16>
__global__ __launch_bounds__(256)
void gemm_wmma(const void* __restrict__ Aop, const float* __restrict__ W,
               const float* __restrict__ bias, void* __restrict__ Cop,
               int M, int K, int N, float scale)
{
    constexpr int BM = 128, BN = 128, BK = 32, LDSW = BK + 8;
    __shared__ unsigned short As[BM][LDSW];   // [m][k]
    __shared__ unsigned short Bs[BN][LDSW];   // transposed: [n][k]

    const int tid  = threadIdx.x;
    const int lane = tid & 31;
    const int wave = tid >> 5;
    const int wm   = wave & 3;     // 4 waves along M (32 rows each)
    const int wn   = wave >> 2;    // 2 waves along N (64 cols each)
    const int m0   = blockIdx.y * BM;
    const int n0   = blockIdx.x * BN;
    const int half = lane >> 4;
    const int lr   = lane & 15;

    v8f acc[2][4];
    #pragma unroll
    for (int i = 0; i < 2; ++i)
        #pragma unroll
        for (int j = 0; j < 4; ++j) acc[i][j] = vzero8();

    const int arow = tid >> 1, acol = (tid & 1) * 16;   // 128 x 32 stage
    const int brow = tid >> 3, bcol = (tid & 7) * 16;   // 32 x 128 stage

    for (int k0 = 0; k0 < K; k0 += BK) {
        // ---- stage A tile
        if (A_BF16) {
            // pure bf16 copy: async DMA straight into LDS, no VGPR round-trip
            const unsigned short* src =
                (const unsigned short*)Aop + (size_t)(m0 + arow) * K + k0 + acol;
            async_copy_b128(src,     &As[arow][acol]);
            async_copy_b128(src + 8, &As[arow][acol + 8]);
        } else {
            const float* src = (const float*)Aop + (size_t)(m0 + arow) * K + k0 + acol;
            #pragma unroll
            for (int j = 0; j < 16; j += 4) {
                f32x4 f = *(const f32x4*)(src + j);
                As[arow][acol + j + 0] = f2bf(f.x);
                As[arow][acol + j + 1] = f2bf(f.y);
                As[arow][acol + j + 2] = f2bf(f.z);
                As[arow][acol + j + 3] = f2bf(f.w);
            }
        }
        // ---- stage B tile transposed (fp32->bf16 convert, so VALU path)
        {
            const float* src = W + (size_t)(k0 + brow) * N + n0 + bcol;
            #pragma unroll
            for (int j = 0; j < 16; j += 4) {
                f32x4 f = *(const f32x4*)(src + j);
                Bs[bcol + j + 0][brow] = f2bf(f.x);
                Bs[bcol + j + 1][brow] = f2bf(f.y);
                Bs[bcol + j + 2][brow] = f2bf(f.z);
                Bs[bcol + j + 3][brow] = f2bf(f.w);
            }
            if (k0 + BK < K) {
                __builtin_prefetch((const void*)(src + (size_t)BK * N), 0, 2);
            }
        }
        if (A_BF16) wait_async_zero();
        __syncthreads();

        // ---- fragments + 8 WMMAs
        FragAB a[2], b[4];
        #pragma unroll
        for (int rt = 0; rt < 2; ++rt) {
            const unsigned short* p = &As[wm * 32 + rt * 16 + lr][0];
            a[rt].h[0] = *(const u16x8*)(p + 16 * half);
            a[rt].h[1] = *(const u16x8*)(p + 16 * half + 8);
        }
        #pragma unroll
        for (int ct = 0; ct < 4; ++ct) {
            const unsigned short* p = &Bs[wn * 64 + ct * 16 + lr][0];
            b[ct].h[0] = *(const u16x8*)(p + 16 * half);
            b[ct].h[1] = *(const u16x8*)(p + 16 * half + 8);
        }
        #pragma unroll
        for (int rt = 0; rt < 2; ++rt)
            #pragma unroll
            for (int ct = 0; ct < 4; ++ct)
                acc[rt][ct] = wmma_bf16(a[rt], b[ct], acc[rt][ct]);
        __syncthreads();
    }

    // ---- epilogue (C layout: row = i + 8*half, col = lane&15)
    #pragma unroll
    for (int rt = 0; rt < 2; ++rt) {
        #pragma unroll
        for (int ct = 0; ct < 4; ++ct) {
            const int col = n0 + wn * 64 + ct * 16 + lr;
            const float bv = bias ? bias[col] : 0.0f;
            #pragma unroll
            for (int i = 0; i < 8; ++i) {
                const int row = m0 + wm * 32 + rt * 16 + i + 8 * half;
                const float v = (acc[rt][ct][i] + bv) * scale;
                if (OUT_BF16)
                    ((unsigned short*)Cop)[(size_t)row * N + col] = f2bf(v);
                else
                    ((float*)Cop)[(size_t)row * N + col] = v;
            }
        }
    }
}

// ================================================================ attention
// One workgroup = (head h, 32 query rows).  Full score rows (<=2048 keys) are
// kept in LDS (32 * 2052 * 4B = 257KB) -- only possible with CDNA5's 320KB WGP
// LDS.  Single pass: WMMA scores -> softmax -> colsum atomics -> WMMA P@V.
__global__ __launch_bounds__(256)
void attn_fused(const unsigned short* __restrict__ Qbf,   // [S][HID] (pre-scaled 1/sqrt(D))
                const unsigned short* __restrict__ Kbf,   // [S][KVLD]
                const unsigned short* __restrict__ Vbf,   // [S][KVLD]
                unsigned short* __restrict__ Obf,         // [S][HID]
                float* __restrict__ curScores)            // [H][S] (atomic accum)
{
    extern __shared__ unsigned char smem[];
    unsigned short* qs = (unsigned short*)smem;            // [32][136]
    unsigned short* vt = qs + 32 * 136;                    // [128][40] (V^T tile)
    float*          sc = (float*)(smem + 18944);           // [32][SC_STRIDE]

    const int tid  = threadIdx.x;
    const int lane = tid & 31;
    const int wave = tid >> 5;
    const int h    = blockIdx.y;
    const int qt   = blockIdx.x;
    const int qbase    = qt * 32;
    const int kvh      = h >> 2;              // N_REP = 4
    const int keyCount = qbase + 32;          // causal horizon for this tile
    const int half = lane >> 4;
    const int lr   = lane & 15;

    // ---- stage Q tile (32 x 128): pure bf16 copy -> async DMA to LDS
    {
        const int r = tid >> 3, dB = (tid & 7) * 16;
        const unsigned short* src = Qbf + (size_t)(qbase + r) * HID + h * Dd + dB;
        async_copy_b128(src,     &qs[r * 136 + dB]);
        async_copy_b128(src + 8, &qs[r * 136 + dB + 8]);
    }
    wait_async_zero();
    __syncthreads();

    // ---- scores: each wave owns 16-key strips; K fragments stream from global
    const int numStrips = keyCount >> 4;
    for (int s = wave; s < numStrips; s += 8) {
        const int colBase = s * 16;
        const int key = colBase + lr;
        v8f acc0 = vzero8(), acc1 = vzero8();
        #pragma unroll
        for (int ds = 0; ds < 4; ++ds) {
            const int d0 = ds * 32;
            FragAB bfr, a0, a1;
            const unsigned short* kp =
                Kbf + (size_t)key * KVLD + kvh * Dd + d0 + 16 * half;
            bfr.h[0] = *(const u16x8*)kp;
            bfr.h[1] = *(const u16x8*)(kp + 8);
            const unsigned short* q0 = &qs[lr * 136 + d0];
            a0.h[0] = *(const u16x8*)(q0 + 16 * half);
            a0.h[1] = *(const u16x8*)(q0 + 16 * half + 8);
            const unsigned short* q1 = &qs[(16 + lr) * 136 + d0];
            a1.h[0] = *(const u16x8*)(q1 + 16 * half);
            a1.h[1] = *(const u16x8*)(q1 + 16 * half + 8);
            acc0 = wmma_bf16(a0, bfr, acc0);
            acc1 = wmma_bf16(a1, bfr, acc1);
        }
        // causal mask while spilling to LDS
        #pragma unroll
        for (int i = 0; i < 8; ++i) {
            const int q0r = i + 8 * half;       // rows 0..15
            const int q1r = 16 + q0r;           // rows 16..31
            sc[q0r * SC_STRIDE + key] = (key <= qbase + q0r) ? acc0[i] : -3.0e38f;
            sc[q1r * SC_STRIDE + key] = (key <= qbase + q1r) ? acc1[i] : -3.0e38f;
        }
    }
    __syncthreads();

    // ---- softmax: each wave normalizes 4 rows
    #pragma unroll
    for (int rr = 0; rr < 4; ++rr) {
        const int row = wave * 4 + rr;
        float* prow = &sc[row * SC_STRIDE];
        float m = -3.0e38f;
        for (int k = lane; k < keyCount; k += 32) m = fmaxf(m, prow[k]);
        #pragma unroll
        for (int off = 16; off > 0; off >>= 1) m = fmaxf(m, __shfl_xor(m, off, 32));
        float l = 0.0f;
        for (int k = lane; k < keyCount; k += 32) {
            const float e = __expf(prow[k] - m);
            prow[k] = e;
            l += e;
        }
        #pragma unroll
        for (int off = 16; off > 0; off >>= 1) l += __shfl_xor(l, off, 32);
        const float inv = 1.0f / l;
        for (int k = lane; k < keyCount; k += 32) prow[k] *= inv;
    }
    __syncthreads();

    // ---- column sums of P -> cur_scores (H2O accumulation)
    for (int k = tid; k < keyCount; k += 256) {
        float s = 0.0f;
        #pragma unroll
        for (int r = 0; r < 32; ++r) s += sc[r * SC_STRIDE + k];
        atomicAdd(&curScores[h * Ss + k], s);
    }

    // ---- P @ V : wave -> row tile (wave&1), col tiles ct0 and ct0+4
    const int rowtile = wave & 1;
    const int ct0 = wave >> 1;        // 0..3
    v8f o0 = vzero8(), o1 = vzero8();
    for (int kk = 0; kk < keyCount; kk += 32) {
        __syncthreads();
        // stage V^T tile (32 keys x 128 d -> vt[d][key]): transpose => VALU path
        {
            const int kr = tid >> 3, dB = (tid & 7) * 16;
            const unsigned short* src =
                Vbf + (size_t)(kk + kr) * KVLD + kvh * Dd + dB;
            u16x8 p0 = *(const u16x8*)src;
            u16x8 p1 = *(const u16x8*)(src + 8);
            #pragma unroll
            for (int j = 0; j < 8; ++j) vt[(dB + j) * 40 + kr]     = p0[j];
            #pragma unroll
            for (int j = 0; j < 8; ++j) vt[(dB + 8 + j) * 40 + kr] = p1[j];
        }
        __syncthreads();
        // A fragment: probabilities (fp32 LDS -> bf16)
        FragAB pa;
        {
            const float* prow = &sc[(rowtile * 16 + lr) * SC_STRIDE + kk];
            #pragma unroll
            for (int j = 0; j < 8; ++j) pa.s[j]     = f2bf(prow[16 * half + j]);
            #pragma unroll
            for (int j = 0; j < 8; ++j) pa.s[8 + j] = f2bf(prow[16 * half + 8 + j]);
        }
        FragAB vb0, vb1;
        {
            const unsigned short* p0 = &vt[(ct0 * 16 + lr) * 40 + 16 * half];
            vb0.h[0] = *(const u16x8*)p0;
            vb0.h[1] = *(const u16x8*)(p0 + 8);
            const unsigned short* p1 = &vt[((ct0 + 4) * 16 + lr) * 40 + 16 * half];
            vb1.h[0] = *(const u16x8*)p1;
            vb1.h[1] = *(const u16x8*)(p1 + 8);
        }
        o0 = wmma_bf16(pa, vb0, o0);
        o1 = wmma_bf16(pa, vb1, o1);
    }
    // ---- store O (bf16, layout (q, h*128+d) for the Wo GEMM)
    #pragma unroll
    for (int i = 0; i < 8; ++i) {
        const int qrow = qbase + rowtile * 16 + i + 8 * half;
        unsigned short* dst = Obf + (size_t)qrow * HID + h * Dd;
        dst[ct0 * 16 + lr]       = f2bf(o0[i]);
        dst[(ct0 + 4) * 16 + lr] = f2bf(o1[i]);
    }
}

// ================================================================ heavy hitter
__global__ __launch_bounds__(256)
void heavy_hitter(const float* __restrict__ curScores,
                  float* __restrict__ maskOut,    // [H][S+1]
                  float* __restrict__ newScores)  // [H][S]
{
    __shared__ float vals[Ss];
    __shared__ float sel[Ss - RECENT];
    __shared__ float msk[Ss + 1];
    __shared__ float rv[256];
    __shared__ int   ri[256];
    const int tid = threadIdx.x;
    const int h   = blockIdx.x;

    for (int k = tid; k < Ss; k += 256) {
        const float v = curScores[h * Ss + k];
        vals[k] = v;
        if (k < Ss - RECENT) sel[k] = v;
    }
    for (int k = tid; k < Ss + 1; k += 256)
        msk[k] = (k >= Ss + 1 - RECENT) ? 1.0f : 0.0f;
    __syncthreads();

    for (int it = 0; it < HEAVY; ++it) {
        float best = -3.0e38f; int bi = Ss;
        for (int k = tid; k < Ss - RECENT; k += 256) {
            const float v = sel[k];
            if (v > best) { best = v; bi = k; }
        }
        rv[tid] = best; ri[tid] = bi;
        __syncthreads();
        for (int s2 = 128; s2 > 0; s2 >>= 1) {
            if (tid < s2) {
                if (rv[tid + s2] > rv[tid] ||
                    (rv[tid + s2] == rv[tid] && ri[tid + s2] < ri[tid])) {
                    rv[tid] = rv[tid + s2]; ri[tid] = ri[tid + s2];
                }
            }
            __syncthreads();
        }
        if (tid == 0) { const int b = ri[0]; msk[b] = 1.0f; sel[b] = -3.0e38f; }
        __syncthreads();
    }

    for (int k = tid; k < Ss + 1; k += 256) maskOut[h * (Ss + 1) + k] = msk[k];
    for (int k = tid; k < Ss; k += 256) {
        const float sm = (k >= Ss - RECENT) ? 1.0f : msk[k];
        newScores[h * Ss + k] = vals[k] * sm;
    }
}

// ================================================================ launcher
extern "C" void kernel_launch(void* const* d_in, const int* in_sizes, int n_in,
                              void* d_out, int out_size, void* d_ws, size_t ws_size,
                              hipStream_t stream) {
    (void)in_sizes; (void)n_in; (void)out_size; (void)ws_size;
    const float* hidden = (const float*)d_in[0];
    // d_in[1] = attention_mask: pure causal; reconstructed analytically.
    const float* Wq = (const float*)d_in[2];
    const float* bq = (const float*)d_in[3];
    const float* Wk = (const float*)d_in[4];
    const float* bk = (const float*)d_in[5];
    const float* Wv = (const float*)d_in[6];
    const float* bv = (const float*)d_in[7];
    const float* Wo = (const float*)d_in[8];

    unsigned char* ws = (unsigned char*)d_ws;
    unsigned short* Qbf = (unsigned short*)ws;  ws += (size_t)Ss * HID * 2;   // 16 MB
    unsigned short* Kbf = (unsigned short*)ws;  ws += (size_t)Ss * KVLD * 2;  //  4 MB
    unsigned short* Vbf = (unsigned short*)ws;  ws += (size_t)Ss * KVLD * 2;  //  4 MB
    unsigned short* Obf = (unsigned short*)ws;  ws += (size_t)Ss * HID * 2;   // 16 MB
    float* curScores    = (float*)ws;                                         // 256 KB

    hipMemsetAsync(curScores, 0, (size_t)Hn * Ss * sizeof(float), stream);

    const dim3 blk(256);
    const float qscale = 0.08838834764831845f;   // 1/sqrt(128)

    gemm_wmma<false, true><<<dim3(HID / 128, Ss / 128), blk, 0, stream>>>(
        hidden, Wq, bq, Qbf, Ss, HID, HID, qscale);
    gemm_wmma<false, true><<<dim3(KVLD / 128, Ss / 128), blk, 0, stream>>>(
        hidden, Wk, bk, Kbf, Ss, HID, KVLD, 1.0f);
    gemm_wmma<false, true><<<dim3(KVLD / 128, Ss / 128), blk, 0, stream>>>(
        hidden, Wv, bv, Vbf, Ss, HID, KVLD, 1.0f);

    const size_t attnLds = 18944 + (size_t)32 * SC_STRIDE * 4;  // 281600 B
    hipFuncSetAttribute((const void*)attn_fused,
                        hipFuncAttributeMaxDynamicSharedMemorySize, (int)attnLds);
    attn_fused<<<dim3(Ss / 32, Hn), blk, attnLds, stream>>>(
        Qbf, Kbf, Vbf, Obf, curScores);

    gemm_wmma<true, false><<<dim3(HID / 128, Ss / 128), blk, 0, stream>>>(
        Obf, Wo, nullptr, d_out, Ss, HID, HID, 1.0f);

    float* maskOut = (float*)d_out + (size_t)Ss * HID;        // 8,388,608
    float* newOut  = maskOut + (size_t)Hn * (Ss + 1);         // +65,568
    heavy_hitter<<<dim3(Hn), blk, 0, stream>>>(curScores, maskOut, newOut);
}
